// GCNModel_66907000537708
// MI455X (gfx1250) — matrix operand order repaired and checked by reading, not compiled, not run on hardware
//
#include <hip/hip_runtime.h>
#include <hip/hip_bf16.h>
#include <math.h>

// ---------------- constants (match reference) ----------------
constexpr int NN  = 50000;   // nodes
constexpr int NE  = 800000;  // edges
constexpr int NG  = 512;     // graphs
constexpr int DIN = 128;
constexpr int DH  = 256;
constexpr int DE  = 128;
constexpr float BN_EPS = 1e-5f;

// ---------------- small helpers ----------------
__device__ __forceinline__ void atomAddF(float* p, float v) {
  __hip_atomic_fetch_add(p, v, __ATOMIC_RELAXED, __HIP_MEMORY_SCOPE_AGENT);
}
__device__ __forceinline__ void atomMaxF(float* p, float v) {
  __hip_atomic_fetch_max(p, v, __ATOMIC_RELAXED, __HIP_MEMORY_SCOPE_AGENT);
}
__device__ __forceinline__ unsigned short f2bf(float f) {
  unsigned int u = __float_as_uint(f);
  return (unsigned short)((u + 0x7FFFu + ((u >> 16) & 1u)) >> 16); // RNE
}

// ---------------- WMMA types ----------------
typedef __bf16 bf16x16 __attribute__((ext_vector_type(16)));
typedef float  f32x8   __attribute__((ext_vector_type(8)));

union BfFrag {
  bf16x16 v;
  uint4          q[2];
  unsigned int   u[8];
  unsigned short s[16];
};

// ---------------- elementwise / setup kernels ----------------
__global__ void k_fill_f32(float* __restrict__ p, float val, size_t n) {
  size_t t = (size_t)blockIdx.x * blockDim.x + threadIdx.x;
  if (t < n) p[t] = val;
}

__global__ void k_edge_deg(const int* __restrict__ ei, float* __restrict__ deg, int E) {
  int e = blockIdx.x * blockDim.x + threadIdx.x;
  if (e < E) atomAddF(&deg[ei[E + e]], 1.0f);   // dst = ei[1][e]
}

__global__ void k_dinv(const float* __restrict__ deg, float* __restrict__ dinv,
                       float* __restrict__ invd, int n) {
  int i = blockIdx.x * blockDim.x + threadIdx.x;
  if (i < n) { float d = deg[i]; dinv[i] = rsqrtf(d); invd[i] = 1.0f / d; }
}

__global__ void k_edge_norm(const int* __restrict__ ei, const float* __restrict__ dinv,
                            float* __restrict__ nrm, int E) {
  int e = blockIdx.x * blockDim.x + threadIdx.x;
  if (e < E) nrm[e] = dinv[ei[e]] * dinv[ei[E + e]];
}

__global__ void k_cast_bf16(const float* __restrict__ x, unsigned short* __restrict__ y,
                            size_t n) {
  size_t t = (size_t)blockIdx.x * blockDim.x + threadIdx.x;
  if (t < n) y[t] = f2bf(x[t]);
}

// ---------------- WMMA GEMM: C[M,N] f32 = A[M,K] bf16 @ B[K,N] bf16 --------
// 256 threads = 8 waves. blockIdx.x = N column tile (16 cols), blockIdx.y =
// group of 8 M row tiles (one per wave). The block's B slab [K x 16] is staged
// in LDS pre-swizzled into the WMMA B-operand per-lane layout, so each wave
// reads its B fragment with two ds_load_b128 per k-step. A loads coalesce to
// global_load_b128 pairs. M tiles exact (50000 = 3125*16); K in {128,256}.
__global__ void __launch_bounds__(256)
k_gemm_bf16(const unsigned short* __restrict__ A,
            const unsigned short* __restrict__ B,
            float* __restrict__ C, int M, int N, int K) {
  __shared__ unsigned short ldsB[256 * 16];   // up to K=256: [ks][lane][idx]

  const int n0 = blockIdx.x << 4;

  // ---- cooperative fill of B slab into fragment layout ----
  // element (k, col): K index k in [0,K), col in [0,16)
  for (int t = threadIdx.x; t < K * 16; t += blockDim.x) {
    int k    = t >> 4;
    int col  = t & 15;
    int ks   = k >> 5;          // k-step
    int kin  = k & 31;
    int half = (kin >> 3) & 1;  // lanes 16-31 hold K=8..15,24..31 of the step
    int idx  = (kin & 7) | ((kin >> 4) << 3);
    int lane = (half << 4) | col;
    ldsB[(((ks << 5) | lane) << 4) | idx] = B[(size_t)k * N + n0 + col];
  }
  __syncthreads();

  const int lane   = threadIdx.x & 31;
  const int waveId = threadIdx.x >> 5;
  const int tm     = blockIdx.y * 8 + waveId;
  const int Mtiles = M >> 4;
  if (tm >= Mtiles) return;                  // wave-uniform -> EXEC all-ones

  const int half = lane >> 4;
  const int l16  = lane & 15;
  const int m0   = tm << 4;

  f32x8 acc = {};
  const int nks = K >> 5;
  for (int ks = 0; ks < nks; ++ks) {
    const int k0 = ks << 5;
    BfFrag a, b;
    // A 16x32 bf16: lane half selects K+{0..7,16..23} vs K+{8..15,24..31}
    const unsigned short* Arow = A + (size_t)(m0 + l16) * K + k0 + half * 8;
#pragma unroll
    for (int v = 0; v < 8; ++v) {
      int kb = ((v & 3) << 1) + ((v >> 2) << 4);      // 0,2,4,6,16,18,20,22
      a.u[v] = *(const unsigned int*)(Arow + kb);
    }
    // B fragment: contiguous 32B per lane in LDS
    const uint4* bp = (const uint4*)(ldsB + (((ks << 5) | lane) << 4));
    b.q[0] = bp[0];
    b.q[1] = bp[1];

    acc = __builtin_amdgcn_wmma_f32_16x16x32_bf16(false, a.v, false, b.v,
                                                  (short)0, acc, false, false);
  }
  // C 16x16 f32: VGPR r -> M = r + half*8, N = n0 + l16
  float* Crow = C + (size_t)(m0 + half * 8) * N + n0 + l16;
#pragma unroll
  for (int r = 0; r < 8; ++r) Crow[(size_t)r * N] = acc[r];
}

// ---------------- edge aggregation: agg[dst] += h[src] * norm[e] -----------
// one thread per (edge, 4-float chunk)
__global__ void k_edge_msg(const int* __restrict__ ei, const float* __restrict__ nrm,
                           const float* __restrict__ h, float* __restrict__ agg,
                           int E, int Dq /* = D/4 */) {
  size_t t = (size_t)blockIdx.x * blockDim.x + threadIdx.x;
  size_t total = (size_t)E * Dq;
  if (t >= total) return;
  int e = (int)(t / Dq);
  int c = (int)(t - (size_t)e * Dq);
  int s = ei[e], d = ei[E + e];
  float norm = nrm[e];
  const float4 v = ((const float4*)(h + (size_t)s * (Dq << 2)))[c];
  float* ad = agg + (size_t)d * (Dq << 2) + (c << 2);
  atomAddF(ad + 0, v.x * norm);
  atomAddF(ad + 1, v.y * norm);
  atomAddF(ad + 2, v.z * norm);
  atomAddF(ad + 3, v.w * norm);
}

// ---------------- agg = relu(agg + h*invdeg + bias)  (in place) ------------
__global__ void k_combine_relu(float* __restrict__ agg, const float* __restrict__ h,
                               const float* __restrict__ invd,
                               const float* __restrict__ bias, int n, int D) {
  size_t t = (size_t)blockIdx.x * blockDim.x + threadIdx.x;
  size_t total = (size_t)n * D;
  if (t >= total) return;
  size_t i = t / D;
  int d = (int)(t - i * D);
  float v = agg[t] + h[t] * invd[i] + bias[d];
  agg[t] = fmaxf(v, 0.0f);
}

// ---------------- batch-norm stats: per-channel sum & sumsq ----------------
// blockDim.x == D; each block covers `rows` nodes; coalesced channel access.
__global__ void k_bn_stats(const float* __restrict__ h, float* __restrict__ sums,
                           float* __restrict__ sumsq, int n, int D, int rows) {
  int d  = threadIdx.x;
  int r0 = blockIdx.x * rows;
  int r1 = r0 + rows; if (r1 > n) r1 = n;
  float s = 0.f, q = 0.f;
  for (int r = r0; r < r1; ++r) {
    float v = h[(size_t)r * D + d];
    s += v; q += v * v;
  }
  atomAddF(&sums[d], s);
  atomAddF(&sumsq[d], q);
}

__global__ void k_bn_finalize(const float* __restrict__ sums, const float* __restrict__ sumsq,
                              const float* __restrict__ gamma, const float* __restrict__ beta,
                              float* __restrict__ scale, float* __restrict__ shift,
                              int n, int D) {
  int d = blockIdx.x * blockDim.x + threadIdx.x;
  if (d >= D) return;
  float inv_n = 1.0f / (float)n;
  float mean = sums[d] * inv_n;
  float var  = sumsq[d] * inv_n - mean * mean;
  float sc   = gamma[d] * rsqrtf(var + BN_EPS);
  scale[d] = sc;
  shift[d] = beta[d] - mean * sc;
}

__global__ void k_bn_apply_bf16(const float* __restrict__ h, unsigned short* __restrict__ y,
                                const float* __restrict__ scale, const float* __restrict__ shift,
                                size_t total, int D) {
  size_t t = (size_t)blockIdx.x * blockDim.x + threadIdx.x;
  if (t >= total) return;
  int d = (int)(t % D);
  y[t] = f2bf(h[t] * scale[d] + shift[d]);
}

__global__ void k_bn_apply_f32(float* __restrict__ h, const float* __restrict__ scale,
                               const float* __restrict__ shift, size_t total, int D) {
  size_t t = (size_t)blockIdx.x * blockDim.x + threadIdx.x;
  if (t >= total) return;
  int d = (int)(t % D);
  h[t] = h[t] * scale[d] + shift[d];
}

// ---------------- segment max into out[512, DE] ----------------------------
__global__ void k_seg_max(const float* __restrict__ h, const int* __restrict__ batch,
                          float* __restrict__ out, int n, int D) {
  size_t t = (size_t)blockIdx.x * blockDim.x + threadIdx.x;
  size_t total = (size_t)n * D;
  if (t >= total) return;
  size_t i = t / D;
  int d = (int)(t - i * D);
  atomMaxF(&out[(size_t)batch[i] * D + d], h[t]);
}

// ---------------- launch ----------------------------------------------------
static inline int cdiv(size_t a, int b) { return (int)((a + b - 1) / b); }

extern "C" void kernel_launch(void* const* d_in, const int* in_sizes, int n_in,
                              void* d_out, int out_size, void* d_ws, size_t ws_size,
                              hipStream_t stream) {
  const float* x      = (const float*)d_in[0];
  const int*   ei     = (const int*)  d_in[1];   // [2, E]
  const int*   batch  = (const int*)  d_in[2];
  const float* W1     = (const float*)d_in[3];
  const float* b1     = (const float*)d_in[4];
  const float* gamma1 = (const float*)d_in[5];
  const float* beta1  = (const float*)d_in[6];
  const float* W2     = (const float*)d_in[7];
  const float* b2     = (const float*)d_in[8];
  const float* gamma2 = (const float*)d_in[9];
  const float* beta2  = (const float*)d_in[10];
  float* out = (float*)d_out;

  const int E = in_sizes[1] / 2;
  const int n = in_sizes[2];   // NN

  // ---- workspace carve-out (256B aligned) ----
  char* w = (char*)d_ws;
  size_t off = 0;
  auto alloc = [&](size_t bytes) -> void* {
    void* p = w + off;
    off += (bytes + 255) & ~(size_t)255;
    return p;
  };
  float*          deg    = (float*)alloc((size_t)n * 4);
  float*          dinv   = (float*)alloc((size_t)n * 4);
  float*          invd   = (float*)alloc((size_t)n * 4);
  float*          enorm  = (float*)alloc((size_t)E * 4);
  unsigned short* xb     = (unsigned short*)alloc((size_t)n * DIN * 2);
  unsigned short* wb1    = (unsigned short*)alloc((size_t)DIN * DH * 2);
  unsigned short* wb2    = (unsigned short*)alloc((size_t)DH * DE * 2);
  float*          hbuf   = (float*)alloc((size_t)n * DH * 4);   // gemm out (both layers)
  float*          aggbuf = (float*)alloc((size_t)n * DH * 4);   // agg / BN in-place
  unsigned short* hbn1b  = (unsigned short*)alloc((size_t)n * DH * 2);
  float*          stats  = (float*)alloc(1024 * 4);             // sums|sumsq|scale|shift
  float* sums  = stats;
  float* sumsq = stats + 256;
  float* scale = stats + 512;
  float* shift = stats + 768;
  (void)ws_size; (void)n_in;

  const int T = 256;

  // ---- degree with self-loop, dinv, 1/deg, per-edge norm ----
  k_fill_f32<<<cdiv(n, T), T, 0, stream>>>(deg, 1.0f, (size_t)n);
  k_edge_deg<<<cdiv(E, T), T, 0, stream>>>(ei, deg, E);
  k_dinv<<<cdiv(n, T), T, 0, stream>>>(deg, dinv, invd, n);
  k_edge_norm<<<cdiv(E, T), T, 0, stream>>>(ei, dinv, enorm, E);

  // ---- bf16 casts ----
  k_cast_bf16<<<cdiv((size_t)n * DIN, T), T, 0, stream>>>(x,  xb,  (size_t)n * DIN);
  k_cast_bf16<<<cdiv((size_t)DIN * DH, T), T, 0, stream>>>(W1, wb1, (size_t)DIN * DH);
  k_cast_bf16<<<cdiv((size_t)DH * DE, T), T, 0, stream>>>(W2, wb2, (size_t)DH * DE);

  // =================== layer 1 ===================
  {
    dim3 grid(DH / 16, cdiv(n / 16, 8));
    k_gemm_bf16<<<grid, 256, 0, stream>>>(xb, wb1, hbuf, n, DH, DIN);

    k_fill_f32<<<cdiv((size_t)n * DH, T), T, 0, stream>>>(aggbuf, 0.0f, (size_t)n * DH);
    k_edge_msg<<<cdiv((size_t)E * (DH / 4), T), T, 0, stream>>>(ei, enorm, hbuf, aggbuf, E, DH / 4);
    k_combine_relu<<<cdiv((size_t)n * DH, T), T, 0, stream>>>(aggbuf, hbuf, invd, b1, n, DH);

    k_fill_f32<<<cdiv((size_t)512, T), T, 0, stream>>>(stats, 0.0f, 512);
    k_bn_stats<<<cdiv(n, 128), DH, 0, stream>>>(aggbuf, sums, sumsq, n, DH, 128);
    k_bn_finalize<<<1, DH, 0, stream>>>(sums, sumsq, gamma1, beta1, scale, shift, n, DH);
    k_bn_apply_bf16<<<cdiv((size_t)n * DH, T), T, 0, stream>>>(aggbuf, hbn1b, scale, shift,
                                                               (size_t)n * DH, DH);
  }

  // =================== layer 2 ===================
  {
    dim3 grid(DE / 16, cdiv(n / 16, 8));
    k_gemm_bf16<<<grid, 256, 0, stream>>>(hbn1b, wb2, hbuf, n, DE, DH);

    k_fill_f32<<<cdiv((size_t)n * DE, T), T, 0, stream>>>(aggbuf, 0.0f, (size_t)n * DE);
    k_edge_msg<<<cdiv((size_t)E * (DE / 4), T), T, 0, stream>>>(ei, enorm, hbuf, aggbuf, E, DE / 4);
    k_combine_relu<<<cdiv((size_t)n * DE, T), T, 0, stream>>>(aggbuf, hbuf, invd, b2, n, DE);

    k_fill_f32<<<cdiv((size_t)512, T), T, 0, stream>>>(stats, 0.0f, 512);
    k_bn_stats<<<cdiv(n, 128), DE, 0, stream>>>(aggbuf, sums, sumsq, n, DE, 128);
    k_bn_finalize<<<1, DE, 0, stream>>>(sums, sumsq, gamma2, beta2, scale, shift, n, DE);
    k_bn_apply_f32<<<cdiv((size_t)n * DE, T), T, 0, stream>>>(aggbuf, scale, shift,
                                                              (size_t)n * DE, DE);
  }

  // =================== segment max ===================
  k_fill_f32<<<cdiv((size_t)NG * DE, T), T, 0, stream>>>(out, -INFINITY, (size_t)NG * DE);
  k_seg_max<<<cdiv((size_t)n * DE, T), T, 0, stream>>>(aggbuf, batch, out, n, DE);
}
